// PoseDecoder_36120674960064
// MI455X (gfx1250) — matrix operand order repaired and checked by reading, not compiled, not run on hardware
//
#include <hip/hip_runtime.h>

// CDNA5 (gfx1250) persistent-recurrence LSTM pose decoder.
// T=256, V=1024, D=H=128, L=128. 64 workgroups x 512 threads (16 wave32).
// Each workgroup owns 16 rows of V and runs the whole T-loop; weights live in
// VGPRs as bf16 WMMA B-fragments; per step: 16x v_wmma_f32_16x16x32_bf16 per
// wave, LDS-staged gates, scalar LSTM update with fp32 c/h/pose in registers.

typedef __attribute__((ext_vector_type(16))) __bf16 v16bf;
typedef __attribute__((ext_vector_type(8)))  float  v8f;

#define TSTEPS 256
#define VROWS  1024
#define DH     128
#define KIN    256        // GEMM K = [pose(128), h(128)]
#define NG     512        // 4*H gate columns
#define XROWB  528        // bytes per x row: 16B aligned, bank-conflict-free
#define GROWF  516        // floats per gates row (padded)

union AFrag { uint4 q[2]; v16bf v; };
union BFrag { unsigned short s[16]; v16bf v; };

__device__ __forceinline__ unsigned short f2bf(float f) {
  unsigned int u = __float_as_uint(f);
  u += 0x7FFFu + ((u >> 16) & 1u);          // round-to-nearest-even
  return (unsigned short)(u >> 16);
}
__device__ __forceinline__ float sigm(float x) {
  return __builtin_amdgcn_rcpf(1.f + __builtin_amdgcn_exp2f(-1.4426950408889634f * x));
}
__device__ __forceinline__ float tanh_f(float x) {
  // tanh(x) = 1 - 2/(1+e^{2x})
  return 1.f - 2.f * __builtin_amdgcn_rcpf(1.f + __builtin_amdgcn_exp2f(2.8853900817779634f * x));
}

__global__ __launch_bounds__(512) void lstm_pose_kernel(
    const float* __restrict__ hid,   const float* __restrict__ box,
    const float* __restrict__ latent,const float* __restrict__ Wih,
    const float* __restrict__ Whh,   const float* __restrict__ bih,
    const float* __restrict__ bhh,   float* __restrict__ out)
{
  __shared__ __align__(16) unsigned char lds_x[16 * XROWB]; // bf16 [pose|h] rows
  __shared__ __align__(16) float         lds_g[16 * GROWF]; // fp32 gates 16x512

  const int tid  = threadIdx.x;
  const int lane = tid & 31;
  const int w    = tid >> 5;             // wave id 0..15 (also update-phase row)
  const int r0   = blockIdx.x * 16;      // first V row of this workgroup
  const int vg   = r0 + w;               // this thread's owned row
  const int d4   = lane * 4;             // this thread's owned 4 cols (update)
  const int ncol = lane & 15;
  const int half = (lane >= 16) ? 16 : 0;
  const int chi  = (lane >= 16) ? 8  : 0;
  const int nb   = w * 32;               // wave's first gate column

  // ---------- one-time: weights -> bf16 B fragments in VGPRs ----------
  // B(32x16) layout: lane<16: col=lane, K=kt*32+0..15; lane>=16: col=lane-16, K=+16..31
  v16bf Bf[2][8];
  #pragma unroll
  for (int tt = 0; tt < 2; ++tt) {
    const int n = nb + tt * 16 + ncol;
    #pragma unroll
    for (int kt = 0; kt < 8; ++kt) {
      const int kg = kt * 32 + half;     // first of 16 consecutive k
      const float* src = (kg < 128) ? (Wih + (size_t)n * KIN + kg)        // pose part
                                    : (Whh + (size_t)n * DH  + (kg - 128)); // h part
      BFrag b;
      #pragma unroll
      for (int j = 0; j < 16; ++j) b.s[j] = f2bf(src[j]);
      Bf[tt][kt] = b.v;
    }
  }

  // ---------- one-time: Lbias[v][n] = latent@Wih[:,128:].T + b_ih + b_hh ----------
  {
    float acc[16];
    const int n0 = lane * 16;
    #pragma unroll
    for (int j = 0; j < 16; ++j) acc[j] = bih[n0 + j] + bhh[n0 + j];
    for (int k = 0; k < 128; ++k) {
      const float lat = latent[(size_t)vg * 128 + k];
      #pragma unroll
      for (int j = 0; j < 16; ++j)
        acc[j] += lat * Wih[(size_t)(n0 + j) * KIN + 128 + k];
    }
    #pragma unroll
    for (int j = 0; j < 16; ++j) lds_g[w * GROWF + n0 + j] = acc[j];
  }

  // ---------- one-time: init h0=c0=hid, pose0=box[0]; stage x=[pose,h] bf16 ----------
  float creg[4], hreg[4], preg[4];
  {
    const float4 h4 = *(const float4*)(hid + (size_t)vg * DH + d4);
    const float4 p4 = *(const float4*)(box + (size_t)vg * DH + d4);
    hreg[0]=h4.x; hreg[1]=h4.y; hreg[2]=h4.z; hreg[3]=h4.w;
    creg[0]=h4.x; creg[1]=h4.y; creg[2]=h4.z; creg[3]=h4.w;
    preg[0]=p4.x; preg[1]=p4.y; preg[2]=p4.z; preg[3]=p4.w;
    uint2 pp, hh;
    pp.x = f2bf(preg[0]) | ((unsigned)f2bf(preg[1]) << 16);
    pp.y = f2bf(preg[2]) | ((unsigned)f2bf(preg[3]) << 16);
    hh.x = f2bf(hreg[0]) | ((unsigned)f2bf(hreg[1]) << 16);
    hh.y = f2bf(hreg[2]) | ((unsigned)f2bf(hreg[3]) << 16);
    *(uint2*)(lds_x + w * XROWB +       d4 * 2) = pp;
    *(uint2*)(lds_x + w * XROWB + 256 + d4 * 2) = hh;
  }
  __syncthreads();

  // Lbias C-fragments (each wave reads only its own 32 columns -> no race later)
  v8f lb0, lb1;
  #pragma unroll
  for (int r = 0; r < 8; ++r) {
    lb0[r] = lds_g[(r + chi) * GROWF + nb + ncol];
    lb1[r] = lds_g[(r + chi) * GROWF + nb + 16 + ncol];
  }

  float* const outPose = out + 2 * (VROWS * DH);

  for (int t = 0; t < TSTEPS; ++t) {
    // ---- GEMM: gates(16x32 cols of this wave) = x(16x256) @ Wc.T, C init = Lbias ----
    v8f a0 = lb0, a1 = lb1;
    #pragma unroll
    for (int kt = 0; kt < 8; ++kt) {
      AFrag a;  // A 16x32 bf16: lane<16 row=lane K{0-15}; lane>=16 K{16-31}
      const unsigned char* p = lds_x + ncol * XROWB + kt * 64 + half * 2;
      a.q[0] = *(const uint4*)p;
      a.q[1] = *(const uint4*)(p + 16);
      a0 = __builtin_amdgcn_wmma_f32_16x16x32_bf16(false, a.v, false, Bf[0][kt],
                                                   (short)0, a0, false, false);
      a1 = __builtin_amdgcn_wmma_f32_16x16x32_bf16(false, a.v, false, Bf[1][kt],
                                                   (short)0, a1, false, false);
    }
    #pragma unroll
    for (int r = 0; r < 8; ++r) {       // C layout: VGPR r -> M=r (+8 for hi lanes)
      lds_g[(r + chi) * GROWF + nb + ncol]      = a0[r];
      lds_g[(r + chi) * GROWF + nb + 16 + ncol] = a1[r];
    }
    __syncthreads();

    // ---- LSTM update: thread owns (row=w, cols d4..d4+3); gate order i,f,g,o ----
    const float4 gi = *(const float4*)&lds_g[w * GROWF +   0 + d4];
    const float4 gf = *(const float4*)&lds_g[w * GROWF + 128 + d4];
    const float4 gg = *(const float4*)&lds_g[w * GROWF + 256 + d4];
    const float4 go = *(const float4*)&lds_g[w * GROWF + 384 + d4];
    const float iv[4] = {gi.x, gi.y, gi.z, gi.w};
    const float fv[4] = {gf.x, gf.y, gf.z, gf.w};
    const float gv[4] = {gg.x, gg.y, gg.z, gg.w};
    const float ov[4] = {go.x, go.y, go.z, go.w};
    #pragma unroll
    for (int j = 0; j < 4; ++j) {
      const float i_ = sigm(iv[j]);
      const float f_ = sigm(fv[j]);
      const float g_ = tanh_f(gv[j]);
      const float o_ = sigm(ov[j]);
      creg[j] = f_ * creg[j] + i_ * g_;
      hreg[j] = o_ * tanh_f(creg[j]);
      preg[j] += hreg[j];
    }
    // emit pose_t (coalesced 512B/wave)
    *(float4*)(outPose + ((size_t)t * VROWS + vg) * DH + d4) =
        make_float4(preg[0], preg[1], preg[2], preg[3]);
    // restage x = [pose, h] as bf16 for next step
    uint2 pp, hh;
    pp.x = f2bf(preg[0]) | ((unsigned)f2bf(preg[1]) << 16);
    pp.y = f2bf(preg[2]) | ((unsigned)f2bf(preg[3]) << 16);
    hh.x = f2bf(hreg[0]) | ((unsigned)f2bf(hreg[1]) << 16);
    hh.y = f2bf(hreg[2]) | ((unsigned)f2bf(hreg[3]) << 16);
    *(uint2*)(lds_x + w * XROWB +       d4 * 2) = pp;
    *(uint2*)(lds_x + w * XROWB + 256 + d4 * 2) = hh;
    __syncthreads();
  }

  // ---- epilogue: hT, cT ----
  *(float4*)(out + (size_t)vg * DH + d4) =
      make_float4(hreg[0], hreg[1], hreg[2], hreg[3]);
  *(float4*)(out + (size_t)VROWS * DH + (size_t)vg * DH + d4) =
      make_float4(creg[0], creg[1], creg[2], creg[3]);
}

extern "C" void kernel_launch(void* const* d_in, const int* in_sizes, int n_in,
                              void* d_out, int out_size, void* d_ws, size_t ws_size,
                              hipStream_t stream) {
  (void)in_sizes; (void)n_in; (void)out_size; (void)d_ws; (void)ws_size;
  const float* hid    = (const float*)d_in[0];
  const float* box    = (const float*)d_in[1];
  const float* latent = (const float*)d_in[2];
  const float* W_ih   = (const float*)d_in[3];
  const float* W_hh   = (const float*)d_in[4];
  const float* b_ih   = (const float*)d_in[5];
  const float* b_hh   = (const float*)d_in[6];
  float* out = (float*)d_out;

  lstm_pose_kernel<<<dim3(VROWS / 16), dim3(512), 0, stream>>>(
      hid, box, latent, W_ih, W_hh, b_ih, b_hh, out);
}
